// LocationSensitiveAttention_69518340653356
// MI455X (gfx1250) — matrix operand order repaired and checked by reading, not compiled, not run on hardware
//
#include <hip/hip_runtime.h>
#include <hip/hip_bf16.h>
#include <math.h>

// ---- shapes ----
#define ATT 128
#define DEC 1024
#define ENC 512
#define LOC 32
#define KSZ 31
#define BB  64
#define TT  2048

#define TBT 128            // t-tile per workgroup
#define WCOLS 544          // 512 (V_w) + 32 (U_w)
#define WST 552            // padded LDS row stride (bf16 units)
#define LST (TBT + 8)      // padded loc row stride (bf16 units)

typedef __attribute__((ext_vector_type(16))) __bf16 v16bf;
typedef __attribute__((ext_vector_type(2)))  __bf16 v2bf;
typedef __attribute__((ext_vector_type(8)))  float  v8f;

union Frag {
    v16bf v;
    unsigned int u[8];
    unsigned short s[16];
};

// native f32->bf16 (compiler selects HW convert, RNE)
__device__ __forceinline__ unsigned short f2bf(float f) {
    __bf16 h = (__bf16)f;
    return __builtin_bit_cast(unsigned short, h);
}
__device__ __forceinline__ unsigned int pack2bf(float f0, float f1) {
    v2bf p;
    p[0] = (__bf16)f0;
    p[1] = (__bf16)f1;
    return __builtin_bit_cast(unsigned int, p);
}

// ---- prep: pack [V_w | U_w] -> bf16 Wcat [128 x 544] ----
__global__ void __launch_bounds__(256) wcat_kernel(const float* __restrict__ Vw,
                                                   const float* __restrict__ Uw,
                                                   unsigned short* __restrict__ Wcat) {
    int i = blockIdx.x * 256 + threadIdx.x;
    if (i >= ATT * WCOLS) return;
    int a = i / WCOLS, c = i % WCOLS;
    float f = (c < ENC) ? Vw[a * ENC + c] : Uw[a * LOC + (c - ENC)];
    Wcat[i] = f2bf(f);
}

// ---- prep: Ws[b,a] = query[b,:] . W_w[a,:] + W_b[a] ----
__global__ void __launch_bounds__(256) ws_kernel(const float* __restrict__ q,
                                                 const float* __restrict__ Ww,
                                                 const float* __restrict__ Wb,
                                                 float* __restrict__ Ws) {
    int idx = blockIdx.x * 256 + threadIdx.x;     // 64*128
    int b = idx >> 7, a = idx & 127;
    const float4* qr = (const float4*)(q + (size_t)b * DEC);
    const float4* wr = (const float4*)(Ww + (size_t)a * DEC);
    float s = 0.f;
    for (int i = 0; i < DEC / 4; i++) {
        float4 x = qr[i], y = wr[i];
        s += x.x * y.x + x.y * y.y + x.z * y.z + x.w * y.w;
    }
    Ws[idx] = s + Wb[a];
}

// ---- main fused kernel: Vh + loc-conv + Uf + tanh + v.dot -> energies ----
__global__ void __launch_bounds__(256) energies_kernel(
    const float* __restrict__ values, const float* __restrict__ cum_attn,
    const float* __restrict__ Fw, const float* __restrict__ vw,
    const unsigned short* __restrict__ WcatG, const float* __restrict__ WsG,
    float* __restrict__ energies) {

    extern __shared__ char smem[];
    unsigned short* sW   = (unsigned short*)smem;            // 128 x WST bf16
    unsigned short* sLoc = sW + ATT * WST;                   // 32 x LST bf16
    float* sWs = (float*)(sLoc + LOC * LST);                 // 128
    float* sVw = sWs + ATT;                                  // 128
    float* sCa = sVw + ATT;                                  // TBT + 30

    const int b  = blockIdx.y;
    const int t0 = blockIdx.x * TBT;
    const int tid = threadIdx.x;

    // stage packed weights (bf16) into padded LDS
    for (int i = tid; i < ATT * WCOLS; i += 256) {
        int a = i / WCOLS, c = i % WCOLS;
        sW[a * WST + c] = WcatG[i];
    }
    if (tid < ATT) { sWs[tid] = WsG[b * ATT + tid]; sVw[tid] = vw[tid]; }
    for (int i = tid; i < TBT + 30; i += 256) {
        int t = t0 - (KSZ / 2) + i;
        sCa[i] = (t >= 0 && t < TT) ? cum_attn[(size_t)b * TT + t] : 0.f;
    }
    __syncthreads();

    // location conv: loc[l,t] = sum_k Fw[l,k] * ca[t0+t-15+k]  -> bf16 LDS
    for (int i = tid; i < LOC * TBT; i += 256) {
        int l = i / TBT, t = i % TBT;
        float s = 0.f;
        #pragma unroll
        for (int k = 0; k < KSZ; k++) s += Fw[l * KSZ + k] * sCa[t + k];
        sLoc[l * LST + t] = f2bf(s);
    }
    __syncthreads();

    const int wave = tid >> 5;           // wave32
    const int lane = tid & 31;
    const int hi = lane >> 4, n = lane & 15;
    const int tw = t0 + wave * 16;

    v8f acc[8] = {};
    const float* vbase = values + (size_t)b * ENC * TT;

    // K over encoder dim, bf16 WMMA 16x16x32, 8 a-blocks
    for (int kc = 0; kc < ENC; kc += 32) {
        Frag Bf;
        const int ko = kc + hi * 16;
        #pragma unroll
        for (int j = 0; j < 8; j++) {
            int e = ko + 2 * j;
            float f0 = vbase[(size_t)e * TT + tw + n];
            float f1 = vbase[(size_t)(e + 1) * TT + tw + n];
            Bf.u[j] = pack2bf(f0, f1);
        }
        #pragma unroll
        for (int ab = 0; ab < 8; ab++) {
            Frag Af;
            const unsigned short* rp = sW + (ab * 16 + n) * WST + kc + hi * 16;
            *(uint4*)&Af.u[0] = *(const uint4*)rp;
            *(uint4*)&Af.u[4] = *(const uint4*)(rp + 8);
            acc[ab] = __builtin_amdgcn_wmma_f32_16x16x32_bf16(
                false, Af.v, false, Bf.v, (short)0, acc[ab], false, false);
        }
    }
    // location term: one more K=32 chunk, A = U_w (Wcat cols 512..543), B = loc tile
    {
        Frag Bf;
        const int ko = hi * 16;
        #pragma unroll
        for (int j = 0; j < 8; j++) {
            int l = ko + 2 * j;
            unsigned short s0 = sLoc[l * LST + wave * 16 + n];
            unsigned short s1 = sLoc[(l + 1) * LST + wave * 16 + n];
            Bf.u[j] = (unsigned)s0 | ((unsigned)s1 << 16);
        }
        #pragma unroll
        for (int ab = 0; ab < 8; ab++) {
            Frag Af;
            const unsigned short* rp = sW + (ab * 16 + n) * WST + ENC + hi * 16;
            *(uint4*)&Af.u[0] = *(const uint4*)rp;
            *(uint4*)&Af.u[4] = *(const uint4*)(rp + 8);
            acc[ab] = __builtin_amdgcn_wmma_f32_16x16x32_bf16(
                false, Af.v, false, Bf.v, (short)0, acc[ab], false, false);
        }
    }

    // epilogue: energies[t] = sum_a vw[a] * tanh(acc + Ws[a])
    float part = 0.f;
    #pragma unroll
    for (int ab = 0; ab < 8; ab++) {
        #pragma unroll
        for (int r = 0; r < 8; r++) {
            int a = ab * 16 + r + 8 * hi;   // C/D layout: rows 0-7 / 8-15 per half
            part += sVw[a] * tanhf(acc[ab][r] + sWs[a]);
        }
    }
    part += __shfl_xor(part, 16, 32);
    if (hi == 0) energies[(size_t)b * TT + tw + n] = part;
}

// ---- softmax over T, in-place (energies -> weights), with mask ----
__global__ void __launch_bounds__(256) softmax_kernel(float* __restrict__ w,
                                                      const unsigned char* __restrict__ mask) {
    __shared__ float red[256];
    const int b = blockIdx.x, tid = threadIdx.x;
    float* row = w + (size_t)b * TT;
    const unsigned char* mrow = mask + (size_t)b * TT;
    float mx = -INFINITY;
    for (int t = tid; t < TT; t += 256) {
        float e = mrow[t] ? -INFINITY : row[t];
        mx = fmaxf(mx, e);
    }
    red[tid] = mx; __syncthreads();
    for (int s = 128; s > 0; s >>= 1) { if (tid < s) red[tid] = fmaxf(red[tid], red[tid + s]); __syncthreads(); }
    mx = red[0]; __syncthreads();
    float sum = 0.f;
    for (int t = tid; t < TT; t += 256) {
        float e = mrow[t] ? -INFINITY : row[t];
        float x = __expf(e - mx);
        row[t] = x; sum += x;
    }
    red[tid] = sum; __syncthreads();
    for (int s = 128; s > 0; s >>= 1) { if (tid < s) red[tid] += red[tid + s]; __syncthreads(); }
    float inv = 1.f / red[0];
    for (int t = tid; t < TT; t += 256) row[t] *= inv;
}

// ---- context[b,e] = sum_t values[b,e,t] * weights[b,t] ----
__global__ void __launch_bounds__(256) context_kernel(const float* __restrict__ values,
                                                      const float* __restrict__ weights,
                                                      float* __restrict__ ctx) {
    __shared__ float sw[TT];
    const int b = blockIdx.y;
    const int e0 = blockIdx.x * 64;
    const int tid = threadIdx.x;
    for (int t = tid; t < TT; t += 256) sw[t] = weights[(size_t)b * TT + t];
    __syncthreads();
    const int wave = tid >> 5, lane = tid & 31;
    const float* vb = values + (size_t)b * ENC * TT;
    for (int i = 0; i < 8; i++) {
        int e = e0 + wave * 8 + i;
        const float* vr = vb + (size_t)e * TT;
        float s = 0.f;
        for (int t = lane; t < TT; t += 32) s += vr[t] * sw[t];
        #pragma unroll
        for (int off = 16; off > 0; off >>= 1) s += __shfl_down(s, off, 32);
        if (lane == 0) ctx[(size_t)b * ENC + e] = s;
    }
}

extern "C" void kernel_launch(void* const* d_in, const int* in_sizes, int n_in,
                              void* d_out, int out_size, void* d_ws, size_t ws_size,
                              hipStream_t stream) {
    const float* query    = (const float*)d_in[0];
    const float* values   = (const float*)d_in[1];
    const float* cum_attn = (const float*)d_in[2];
    const unsigned char* mask = (const unsigned char*)d_in[3];
    const float* W_w = (const float*)d_in[4];
    const float* W_b = (const float*)d_in[5];
    const float* V_w = (const float*)d_in[6];
    const float* U_w = (const float*)d_in[7];
    const float* F_w = (const float*)d_in[8];
    const float* v_w = (const float*)d_in[9];

    float* ctx = (float*)d_out;                 // [64,512]
    float* wts = (float*)d_out + BB * ENC;      // [64,2048]  (energies staged here)

    unsigned short* Wcat = (unsigned short*)d_ws;              // 128*544 bf16 = 139264 B
    float* Ws = (float*)((char*)d_ws + ATT * WCOLS * 2);       // 64*128 f32

    wcat_kernel<<<(ATT * WCOLS + 255) / 256, 256, 0, stream>>>(V_w, U_w, Wcat);
    ws_kernel<<<(BB * ATT) / 256, 256, 0, stream>>>(query, W_w, W_b, Ws);

    size_t smem = (size_t)ATT * WST * 2 + (size_t)LOC * LST * 2
                + (ATT + ATT + (TBT + 32)) * sizeof(float);
    dim3 grid(TT / TBT, BB);
    energies_kernel<<<grid, 256, smem, stream>>>(values, cum_attn, F_w, v_w, Wcat, Ws, wts);

    softmax_kernel<<<BB, 256, 0, stream>>>(wts, mask);
    context_kernel<<<dim3(ENC / 64, BB), 256, 0, stream>>>(values, wts, ctx);
}
